// HierGATBlock_56221121904664
// MI455X (gfx1250) — matrix operand order repaired and checked by reading, not compiled, not run on hardware
//
#include <hip/hip_runtime.h>
#include <hip/hip_bf16.h>
#include <math.h>

typedef __attribute__((ext_vector_type(16))) _Float16 v16h;
typedef __attribute__((ext_vector_type(8)))  float    v8f;
typedef __attribute__((ext_vector_type(4)))  _Float16 h4v;

#define Bv   8
#define Sv   1024
#define Dv   256
#define Ev   4096
#define EDv  16
#define FFv  1024
#define Hv   8
#define DKv  32
#define BSv  (Bv * Sv)

// ---------------------------------------------------------------------------
// WMMA fragment helpers (wave32, 16x16x32 f16 -> f32), contiguous k order in A and B, C/D per ISA 7.12.2.
// A (16x32): lane m = lane&15; lanes 0-15 read k=0..15,
//            lanes 16-31 read k=16..31  -> 16 contiguous halves (same as B).
// B (32x16) from TRANSPOSED LDS (pT[n][k]): lane n = lane&15;
//            lanes 0-15 k=0..15, lanes 16-31 k=16..31 -> 16 contiguous halves.
// C/D (16x16): VGPR i: lanes 0-15 -> M=i, lanes 16-31 -> M=8+i; N = lane&15.
// ---------------------------------------------------------------------------
__device__ inline v8f vzero8() {
  v8f z;
#pragma unroll
  for (int i = 0; i < 8; ++i) z[i] = 0.0f;
  return z;
}

__device__ inline v16h ld_frag_a16(const _Float16* p, int ld) {
  const int lane = threadIdx.x & 31;
  const int m = lane & 15;
  const int kb = (lane >> 4) << 4;   // 0 or 16
  v16h a;
#pragma unroll
  for (int j = 0; j < 8; ++j) {
    a[j]     = p[m * ld + kb + j];        // k = kb + j      (contiguous 8)
    a[8 + j] = p[m * ld + kb + 8 + j];    // k = kb + 8 + j  (contiguous 8)
  }
  return a;
}

__device__ inline v16h ld_frag_bT(const _Float16* pT, int ld) {
  const int lane = threadIdx.x & 31;
  const int n = lane & 15;
  const int kb = (lane >> 4) << 4;   // 0 or 16
  v16h b;
#pragma unroll
  for (int e = 0; e < 16; ++e) b[e] = pT[n * ld + kb + e];  // 16 contiguous
  return b;
}

__device__ inline h4v cvt_h4(float4 f) {
  h4v h;
  h[0] = (_Float16)f.x; h[1] = (_Float16)f.y;
  h[2] = (_Float16)f.z; h[3] = (_Float16)f.w;
  return h;
}

// ---------------------------------------------------------------------------
// Generic GEMM: C[M,N] = act(A[M,K] @ W[K,N] + bias)
// A,W f32 row-major -> f16 LDS tiles (B transposed), f32 WMMA accumulate.
// Double-buffered LDS; register-prefetched global loads (all outstanding
// together); fixed trip-count staging (no exec-mask chains).
// Requires M%128==0, N%128==0, K%32==0 (true for all uses: K = 256 or 1024).
// block = 256 threads (8 waves, 2x4), tile 128x128, K-step 32.
// ---------------------------------------------------------------------------
#define TM 128
#define TN 128
#define TK 32
#define LDA (TK + 8)     // 40 halves -> 80B rows (16B aligned)
#define LDBT (TK + 8)    // BsT[n][k], 40 halves per row

__global__ __launch_bounds__(256)
void gemm_wmma_f16(const float* __restrict__ A, const float* __restrict__ W,
                   const float* __restrict__ bias, float* __restrict__ C,
                   int M, int N, int K, int relu) {
  __shared__ _Float16 As[2][TM * LDA];     // 2 x 10240 B
  __shared__ _Float16 BsT[2][TN * LDBT];   // 2 x 10240 B, transposed [n][k]

  const int bm = blockIdx.x * TM;
  const int bn = blockIdx.y * TN;
  const int tid = threadIdx.x;
  const int wave = tid >> 5, lane = tid & 31;
  const int wr = wave >> 2;          // 0..1  (64 rows each)
  const int wc = wave & 3;           // 0..3  (32 cols each)

  // fixed per-thread staging coordinates (4 chunks of 256 threads each)
  int ar[4], ac[4], br[4], bc[4];
#pragma unroll
  for (int it = 0; it < 4; ++it) {
    const int i = tid + it * 256;
    ar[it] = i >> 3;            // A row   (0..127)
    ac[it] = (i & 7) << 2;      // A col4  (0..28)
    br[it] = i >> 5;            // W row k (0..31)
    bc[it] = (i & 31) << 2;     // W col4  (0..124)
  }

  v8f acc[4][2];
#pragma unroll
  for (int i = 0; i < 4; ++i)
#pragma unroll
    for (int j = 0; j < 2; ++j) acc[i][j] = vzero8();

  float4 ra[4], rb[4];
  // prologue: batch-load K-step 0 (all 8 b128 loads outstanding together)
#pragma unroll
  for (int it = 0; it < 4; ++it) {
    ra[it] = *(const float4*)(A + (size_t)(bm + ar[it]) * K + ac[it]);
    rb[it] = *(const float4*)(W + (size_t)br[it] * N + bn + bc[it]);
  }
#pragma unroll
  for (int it = 0; it < 4; ++it) {
    *(h4v*)&As[0][ar[it] * LDA + ac[it]] = cvt_h4(ra[it]);
    const h4v hb = cvt_h4(rb[it]);
    BsT[0][(bc[it] + 0) * LDBT + br[it]] = hb[0];
    BsT[0][(bc[it] + 1) * LDBT + br[it]] = hb[1];
    BsT[0][(bc[it] + 2) * LDBT + br[it]] = hb[2];
    BsT[0][(bc[it] + 3) * LDBT + br[it]] = hb[3];
  }

  int cur = 0;
  for (int k0 = 0; k0 < K; k0 += TK) {
    __syncthreads();                       // buf[cur] stores visible
    const bool more = (k0 + TK) < K;
    if (more) {                            // prefetch next K-step into regs
      const int kn = k0 + TK;
#pragma unroll
      for (int it = 0; it < 4; ++it) {
        ra[it] = *(const float4*)(A + (size_t)(bm + ar[it]) * K + kn + ac[it]);
        rb[it] = *(const float4*)(W + (size_t)(kn + br[it]) * N + bn + bc[it]);
      }
    }

    // compute on buf[cur] while prefetch loads are in flight
    const v16h bf0 = ld_frag_bT(&BsT[cur][(wc * 32 + 0) * LDBT], LDBT);
    const v16h bf1 = ld_frag_bT(&BsT[cur][(wc * 32 + 16) * LDBT], LDBT);
#pragma unroll
    for (int mi = 0; mi < 4; ++mi) {
      const v16h af = ld_frag_a16(&As[cur][(wr * 64 + mi * 16) * LDA], LDA);
      acc[mi][0] = __builtin_amdgcn_wmma_f32_16x16x32_f16(
          false, af, false, bf0, (short)0, acc[mi][0], false, false);
      acc[mi][1] = __builtin_amdgcn_wmma_f32_16x16x32_f16(
          false, af, false, bf1, (short)0, acc[mi][1], false, false);
    }

    if (more) {                            // store prefetch into buf[cur^1]
      const int nxt = cur ^ 1;
#pragma unroll
      for (int it = 0; it < 4; ++it) {
        *(h4v*)&As[nxt][ar[it] * LDA + ac[it]] = cvt_h4(ra[it]);
        const h4v hb = cvt_h4(rb[it]);
        BsT[nxt][(bc[it] + 0) * LDBT + br[it]] = hb[0];
        BsT[nxt][(bc[it] + 1) * LDBT + br[it]] = hb[1];
        BsT[nxt][(bc[it] + 2) * LDBT + br[it]] = hb[2];
        BsT[nxt][(bc[it] + 3) * LDBT + br[it]] = hb[3];
      }
    }
    cur ^= 1;
  }

  const int nloc = lane & 15;
  const int mbase = (lane >> 4) * 8;
#pragma unroll
  for (int mi = 0; mi < 4; ++mi) {
#pragma unroll
    for (int nj = 0; nj < 2; ++nj) {
      const int gn = bn + wc * 32 + nj * 16 + nloc;
      const float bv = bias ? bias[gn] : 0.0f;
#pragma unroll
      for (int r = 0; r < 8; ++r) {
        const int gm = bm + wr * 64 + mi * 16 + mbase + r;
        float v = acc[mi][nj][r] + bv;
        if (relu) v = fmaxf(v, 0.0f);
        C[(size_t)gm * N + gn] = v;
      }
    }
  }
}

// ---------------------------------------------------------------------------
// Flash attention, dk = 32, query-axis mask (reference masks rows).
// grid = (S/128, H, B), block = 256 (8 waves); wave owns 16 q rows.
// scores: B = K^T -> natural Ks[kv][d] is already the transposed layout.
// P@V:    B = V   -> stage VsT[d][kv].  K/V chunks double-buffered with
// register prefetch (exactly one float4 pair per thread per chunk).
// ---------------------------------------------------------------------------
#define LQK (DKv + 8)    // 40

__global__ __launch_bounds__(256)
void flash_attn_dk32(const float* __restrict__ Q, const float* __restrict__ Kg,
                     const float* __restrict__ Vg,
                     const unsigned char* __restrict__ qmask,
                     float* __restrict__ O, int Skv) {
  __shared__ _Float16 Qs[128 * LQK];          // 10240 B
  __shared__ _Float16 Ks[2][32 * LQK];        // [kv][d]  2 x 2560 B
  __shared__ _Float16 VsT[2][DKv * LQK];      // [d][kv]  2 x 2560 B
  __shared__ _Float16 Ps[8 * 16 * LQK];       // per-wave 16x32 P, stride 40

  const int qt = blockIdx.x, h = blockIdx.y, b = blockIdx.z;
  const int tid = threadIdx.x, wave = tid >> 5, lane = tid & 31;
  const float scale = 0.17677669529663687f;   // 1/sqrt(32)

  // stage Q (pre-scaled): 4 fixed chunks, loads batched
  {
    float4 fq[4];
#pragma unroll
    for (int it = 0; it < 4; ++it) {
      const int i = tid + it * 256;
      const int r = i >> 3, c4 = (i & 7) << 2;
      fq[it] = *(const float4*)(Q + ((size_t)(b * Sv + qt * 128 + r)) * Dv +
                                h * DKv + c4);
    }
#pragma unroll
    for (int it = 0; it < 4; ++it) {
      const int i = tid + it * 256;
      const int r = i >> 3, c4 = (i & 7) << 2;
      h4v hq;
      hq[0] = (_Float16)(fq[it].x * scale); hq[1] = (_Float16)(fq[it].y * scale);
      hq[2] = (_Float16)(fq[it].z * scale); hq[3] = (_Float16)(fq[it].w * scale);
      *(h4v*)&Qs[r * LQK + c4] = hq;
    }
  }

  // per-thread K/V staging coordinates: one float4 pair per 32-row chunk
  const int sr = tid >> 3;            // kv row   (0..31)
  const int sc = (tid & 7) << 2;      // d col4   (0..28)

  const int mbase = (lane >> 4) * 8;
  float mb8[8];
#pragma unroll
  for (int i = 0; i < 8; ++i) {
    const int qrow = qt * 128 + wave * 16 + mbase + i;
    mb8[i] = qmask[b * Sv + qrow] ? -1e9f : 0.0f;
  }

  v8f acc0 = vzero8(), acc1 = vzero8();
  float mrow[8], lrow[8];
#pragma unroll
  for (int i = 0; i < 8; ++i) { mrow[i] = -3.0e38f; lrow[i] = 0.0f; }

  _Float16* Pw = &Ps[wave * 16 * LQK];

  // prologue: load + store chunk 0
  {
    const size_t base = ((size_t)(b * Skv + sr)) * Dv + h * DKv + sc;
    const float4 fk = *(const float4*)(Kg + base);
    const float4 fv = *(const float4*)(Vg + base);
    *(h4v*)&Ks[0][sr * LQK + sc] = cvt_h4(fk);
    const h4v hv = cvt_h4(fv);
    VsT[0][(sc + 0) * LQK + sr] = hv[0];
    VsT[0][(sc + 1) * LQK + sr] = hv[1];
    VsT[0][(sc + 2) * LQK + sr] = hv[2];
    VsT[0][(sc + 3) * LQK + sr] = hv[3];
  }
  __syncthreads();
  const v16h qfrag = ld_frag_a16(&Qs[(wave * 16) * LQK], LQK);

  int cur = 0;
  for (int kv0 = 0; kv0 < Skv; kv0 += 32) {
    __syncthreads();                       // buf[cur] stores visible
    const bool more = (kv0 + 32) < Skv;
    float4 fk, fv;
    if (more) {
      const size_t base = ((size_t)(b * Skv + kv0 + 32 + sr)) * Dv + h * DKv + sc;
      fk = *(const float4*)(Kg + base);
      fv = *(const float4*)(Vg + base);
    }

    const v16h kb0 = ld_frag_bT(&Ks[cur][0], LQK);            // kv cols 0-15
    const v16h kb1 = ld_frag_bT(&Ks[cur][16 * LQK], LQK);     // kv cols 16-31
    v8f s0 = __builtin_amdgcn_wmma_f32_16x16x32_f16(
        false, qfrag, false, kb0, (short)0, vzero8(), false, false);
    v8f s1 = __builtin_amdgcn_wmma_f32_16x16x32_f16(
        false, qfrag, false, kb1, (short)0, vzero8(), false, false);

    // online softmax per row (rows in VGPR index, cols across 16 lanes)
#pragma unroll
    for (int i = 0; i < 8; ++i) {
      const float a0 = s0[i] + mb8[i];
      const float a1 = s1[i] + mb8[i];
      float rmax = fmaxf(a0, a1);
#pragma unroll
      for (int xm = 1; xm < 16; xm <<= 1)
        rmax = fmaxf(rmax, __shfl_xor(rmax, xm, 32));
      const float mnew = fmaxf(mrow[i], rmax);
      const float p0 = __expf(a0 - mnew);
      const float p1 = __expf(a1 - mnew);
      float rsum = p0 + p1;
#pragma unroll
      for (int xm = 1; xm < 16; xm <<= 1) rsum += __shfl_xor(rsum, xm, 32);
      const float corr = __expf(mrow[i] - mnew);
      lrow[i] = lrow[i] * corr + rsum;
      mrow[i] = mnew;
      acc0[i] *= corr;
      acc1[i] *= corr;
      const int row = mbase + i, col = lane & 15;
      Pw[row * LQK + col]      = (_Float16)p0;  // per-wave scratch: DS in-order
      Pw[row * LQK + col + 16] = (_Float16)p1;
    }

    const v16h pfrag = ld_frag_a16(Pw, LQK);
    const v16h vb0 = ld_frag_bT(&VsT[cur][0], LQK);           // d cols 0-15
    const v16h vb1 = ld_frag_bT(&VsT[cur][16 * LQK], LQK);    // d cols 16-31
    acc0 = __builtin_amdgcn_wmma_f32_16x16x32_f16(
        false, pfrag, false, vb0, (short)0, acc0, false, false);
    acc1 = __builtin_amdgcn_wmma_f32_16x16x32_f16(
        false, pfrag, false, vb1, (short)0, acc1, false, false);

    if (more) {                            // store prefetch into buf[cur^1]
      const int nxt = cur ^ 1;
      *(h4v*)&Ks[nxt][sr * LQK + sc] = cvt_h4(fk);
      const h4v hv = cvt_h4(fv);
      VsT[nxt][(sc + 0) * LQK + sr] = hv[0];
      VsT[nxt][(sc + 1) * LQK + sr] = hv[1];
      VsT[nxt][(sc + 2) * LQK + sr] = hv[2];
      VsT[nxt][(sc + 3) * LQK + sr] = hv[3];
    }
    cur ^= 1;
  }

  const int d0 = lane & 15;
#pragma unroll
  for (int i = 0; i < 8; ++i) {
    const int qrow = qt * 128 + wave * 16 + mbase + i;
    const float inv = 1.0f / lrow[i];
    const size_t base = ((size_t)(b * Sv + qrow)) * Dv + h * DKv;
    O[base + d0]      = acc0[i] * inv;
    O[base + d0 + 16] = acc1[i] * inv;
  }
}

// ---------------------------------------------------------------------------
// GAT edge score: one wave per edge; dot([h_src,h_dst,edge_attr], a), leaky-relu
// ---------------------------------------------------------------------------
__global__ __launch_bounds__(256)
void edge_score(const float* __restrict__ xt, const int* __restrict__ eidx,
                const float* __restrict__ eattr,
                const unsigned char* __restrict__ emask,
                const float* __restrict__ a, float* __restrict__ eout) {
  const int tid = threadIdx.x, wave = tid >> 5, lane = tid & 31;
  const int e = blockIdx.x * 8 + wave;      // 0 .. B*E-1
  const int b = e >> 12;                    // E = 4096
  const int src = eidx[(size_t)e * 2 + 0];
  const int dst = eidx[(size_t)e * 2 + 1];
  const float* hs = xt + ((size_t)b * Sv + src) * Dv;
  const float* hd = xt + ((size_t)b * Sv + dst) * Dv;
  float acc = 0.0f;
#pragma unroll
  for (int j = lane; j < Dv; j += 32)
    acc += hs[j] * a[j] + hd[j] * a[Dv + j];
  if (lane < EDv) acc += eattr[(size_t)e * EDv + lane] * a[2 * Dv + lane];
#pragma unroll
  for (int xm = 1; xm < 32; xm <<= 1) acc += __shfl_xor(acc, xm, 32);
  if (lane == 0) {
    const float v = acc > 0.0f ? acc : 0.2f * acc;   // leaky_relu(0.2)
    eout[e] = emask[e] ? -1e30f : v;
  }
}

__global__ __launch_bounds__(256)
void edge_softmax(const float* __restrict__ eout,
                  const unsigned char* __restrict__ emask,
                  float* __restrict__ attn) {
  __shared__ float red[256];
  const int b = blockIdx.x, tid = threadIdx.x;
  const float* ein = eout + (size_t)b * Ev;
  float mx = -3.0e38f;
  for (int i = tid; i < Ev; i += 256) mx = fmaxf(mx, ein[i]);
  red[tid] = mx; __syncthreads();
  for (int s = 128; s > 0; s >>= 1) {
    if (tid < s) red[tid] = fmaxf(red[tid], red[tid + s]);
    __syncthreads();
  }
  mx = red[0]; __syncthreads();
  float sum = 0.0f;
  for (int i = tid; i < Ev; i += 256) sum += __expf(ein[i] - mx);
  red[tid] = sum; __syncthreads();
  for (int s = 128; s > 0; s >>= 1) {
    if (tid < s) red[tid] += red[tid + s];
    __syncthreads();
  }
  const float inv = 1.0f / red[0];
  for (int i = tid; i < Ev; i += 256) {
    const float av = __expf(ein[i] - mx) * inv;
    attn[(size_t)b * Ev + i] = emask[(size_t)b * Ev + i] ? 0.0f : av;
  }
}

__global__ __launch_bounds__(256)
void gat_scatter(const float* __restrict__ xt, const int* __restrict__ eidx,
                 const float* __restrict__ attn, float* __restrict__ agg) {
  const int e = blockIdx.x;
  const int b = e >> 12;
  const int src = eidx[(size_t)e * 2 + 0];
  const int dst = eidx[(size_t)e * 2 + 1];
  const float a = attn[e];
  const int d = threadIdx.x;
  atomicAdd(&agg[((size_t)b * Sv + dst) * Dv + d],
            a * xt[((size_t)b * Sv + src) * Dv + d]);
}

__global__ __launch_bounds__(256)
void zero_f32(float* __restrict__ p, int n) {
  const int i = blockIdx.x * 256 + threadIdx.x;
  if (i < n) p[i] = 0.0f;
}

// zero out rows whose mask bit is set (valid_n multiply); i spans rows*D
__global__ __launch_bounds__(256)
void mask_rows(float* __restrict__ p, const unsigned char* __restrict__ mask,
               int rows) {
  const int i = blockIdx.x * 256 + threadIdx.x;
  const int r = i >> 8;            // D = 256
  if (r < rows && mask[r]) p[i] = 0.0f;
}

// layer_norm(2*x) with affine (the reference's h+h residual), D = 256
__global__ __launch_bounds__(256)
void layer_norm2(const float* __restrict__ in, const float* __restrict__ g,
                 const float* __restrict__ bta, float* __restrict__ out) {
  __shared__ float red[256];
  const int r = blockIdx.x, t = threadIdx.x;
  const float x = 2.0f * in[(size_t)r * Dv + t];
  red[t] = x; __syncthreads();
  for (int s = 128; s > 0; s >>= 1) {
    if (t < s) red[t] += red[t + s];
    __syncthreads();
  }
  const float mean = red[0] * (1.0f / Dv);
  __syncthreads();
  const float d = x - mean;
  red[t] = d * d; __syncthreads();
  for (int s = 128; s > 0; s >>= 1) {
    if (t < s) red[t] += red[t + s];
    __syncthreads();
  }
  const float var = red[0] * (1.0f / Dv);
  out[(size_t)r * Dv + t] = d * rsqrtf(var + 1e-5f) * g[t] + bta[t];
}

// ---------------------------------------------------------------------------
extern "C" void kernel_launch(void* const* d_in, const int* in_sizes, int n_in,
                              void* d_out, int out_size, void* d_ws,
                              size_t ws_size, hipStream_t stream) {
  const float* x        = (const float*)d_in[0];
  const int*   eidx     = (const int*)d_in[1];
  const float* eattr    = (const float*)d_in[2];
  const float* enc      = (const float*)d_in[3];
  const unsigned char* tmask = (const unsigned char*)d_in[4];
  // d_in[5] memory_mask: unused by the reference math
  const unsigned char* emask = (const unsigned char*)d_in[6];
  const float* W_gat = (const float*)d_in[7];
  const float* a_gat = (const float*)d_in[8];
  const float* sa_wq = (const float*)d_in[9];
  const float* sa_wk = (const float*)d_in[10];
  const float* sa_wv = (const float*)d_in[11];
  const float* sa_wo = (const float*)d_in[12];
  const float* ed_wq = (const float*)d_in[13];
  const float* ed_wk = (const float*)d_in[14];
  const float* ed_wv = (const float*)d_in[15];
  const float* ed_wo = (const float*)d_in[16];
  const float* ff_w1 = (const float*)d_in[17];
  const float* ff_b1 = (const float*)d_in[18];
  const float* ff_w2 = (const float*)d_in[19];
  const float* ff_b2 = (const float*)d_in[20];
  const float* ln1_g = (const float*)d_in[21];
  const float* ln1_b = (const float*)d_in[22];
  const float* ln2_g = (const float*)d_in[23];
  const float* ln2_b = (const float*)d_in[24];
  const float* ln3_g = (const float*)d_in[25];
  const float* ln3_b = (const float*)d_in[26];
  const float* ln4_g = (const float*)d_in[27];
  const float* ln4_b = (const float*)d_in[28];
  float* out = (float*)d_out;

  const size_t NM = (size_t)BSv * Dv;   // 2M floats
  float* ws  = (float*)d_ws;
  float* xt  = ws;
  float* agg = xt  + NM;
  float* h1  = agg + NM;
  float* qb  = h1  + NM;
  float* kb  = qb  + NM;
  float* vb  = kb  + NM;
  float* ob  = vb  + NM;
  float* t1  = ob  + NM;
  float* h2  = t1  + NM;
  float* h3  = h2  + NM;
  float* ffb = h3  + NM;                 // 8M floats
  float* eb  = ffb + (size_t)BSv * FFv;  // B*E
  float* ab  = eb  + (size_t)Bv * Ev;

  const dim3 blk(256);
  const dim3 gNN(BSv / TM, Dv / TN);     // (64, 2)  N = 256
  const dim3 gNF(BSv / TM, FFv / TN);    // (64, 8)  N = 1024
  const dim3 gFA(Sv / 128, Hv, Bv);      // (8, 8, 8)

  // ---- GAT ----
  gemm_wmma_f16<<<gNN, blk, 0, stream>>>(x, W_gat, nullptr, xt, BSv, Dv, Dv, 0);
  mask_rows<<<BSv, blk, 0, stream>>>(xt, tmask, BSv);
  edge_score<<<(Bv * Ev) / 8, blk, 0, stream>>>(xt, eidx, eattr, emask, a_gat, eb);
  edge_softmax<<<Bv, blk, 0, stream>>>(eb, emask, ab);
  zero_f32<<<(int)(NM / 256), blk, 0, stream>>>(agg, (int)NM);
  gat_scatter<<<Bv * Ev, blk, 0, stream>>>(xt, eidx, ab, agg);
  mask_rows<<<BSv, blk, 0, stream>>>(agg, tmask, BSv);
  layer_norm2<<<BSv, blk, 0, stream>>>(agg, ln1_g, ln1_b, h1);

  // ---- self attention ----
  gemm_wmma_f16<<<gNN, blk, 0, stream>>>(h1, sa_wq, nullptr, qb, BSv, Dv, Dv, 0);
  gemm_wmma_f16<<<gNN, blk, 0, stream>>>(h1, sa_wk, nullptr, kb, BSv, Dv, Dv, 0);
  gemm_wmma_f16<<<gNN, blk, 0, stream>>>(h1, sa_wv, nullptr, vb, BSv, Dv, Dv, 0);
  flash_attn_dk32<<<gFA, blk, 0, stream>>>(qb, kb, vb, tmask, ob, Sv);
  gemm_wmma_f16<<<gNN, blk, 0, stream>>>(ob, sa_wo, nullptr, t1, BSv, Dv, Dv, 0);
  layer_norm2<<<BSv, blk, 0, stream>>>(t1, ln2_g, ln2_b, h2);

  // ---- cross attention ----
  gemm_wmma_f16<<<gNN, blk, 0, stream>>>(h2, ed_wq, nullptr, qb, BSv, Dv, Dv, 0);
  gemm_wmma_f16<<<gNN, blk, 0, stream>>>(enc, ed_wk, nullptr, kb, BSv, Dv, Dv, 0);
  gemm_wmma_f16<<<gNN, blk, 0, stream>>>(enc, ed_wv, nullptr, vb, BSv, Dv, Dv, 0);
  flash_attn_dk32<<<gFA, blk, 0, stream>>>(qb, kb, vb, tmask, ob, Sv);
  gemm_wmma_f16<<<gNN, blk, 0, stream>>>(ob, ed_wo, nullptr, t1, BSv, Dv, Dv, 0);
  layer_norm2<<<BSv, blk, 0, stream>>>(t1, ln3_g, ln3_b, h3);

  // ---- feed-forward ----
  gemm_wmma_f16<<<gNF, blk, 0, stream>>>(h3, ff_w1, ff_b1, ffb, BSv, FFv, Dv, 1);
  gemm_wmma_f16<<<gNN, blk, 0, stream>>>(ffb, ff_w2, ff_b2, t1, BSv, Dv, FFv, 0);
  layer_norm2<<<BSv, blk, 0, stream>>>(t1, ln4_g, ln4_b, out);
}
